// Attention_40106404610880
// MI455X (gfx1250) — matrix operand order repaired and checked by reading, not compiled, not run on hardware
//
#include <hip/hip_runtime.h>

// ---------------- constants ----------------
static constexpr int DIMc   = 2048;
static constexpr int DH     = 64;
static constexpr int QH     = 32;
static constexpr int KVH    = 8;
static constexpr int Bc     = 2;
static constexpr int Nc     = 2048;
static constexpr int Mrows  = Bc * Nc;      // 4096
static constexpr int NQKV   = 3072;         // 2048 q + 512 k + 512 v

typedef unsigned short u16;
typedef __attribute__((ext_vector_type(8)))  __bf16 bfx8;
typedef __attribute__((ext_vector_type(16))) __bf16 bfx16;
typedef __attribute__((ext_vector_type(8)))  float  v8f;
typedef int v4i __attribute__((vector_size(16)));

// ---------------- CDNA5 async global->LDS path ----------------
#if defined(__HIP_DEVICE_COMPILE__) && defined(__gfx1250__) && \
    __has_builtin(__builtin_amdgcn_global_load_async_to_lds_b128) && \
    __has_builtin(__builtin_amdgcn_s_wait_asynccnt)
#define HAVE_ASYNC 1
#else
#define HAVE_ASYNC 0
#endif

#if HAVE_ASYNC
__device__ __forceinline__ void async_ld_b128(const u16* g, u16* l) {
    __builtin_amdgcn_global_load_async_to_lds_b128(
        (__attribute__((address_space(1))) v4i*)(g),
        (__attribute__((address_space(3))) v4i*)(l),
        0, 0);
}
__device__ __forceinline__ void wait_async() { __builtin_amdgcn_s_wait_asynccnt(0); }
#else
__device__ __forceinline__ void wait_async() {}
#endif

// ---------------- helpers ----------------
__device__ __forceinline__ u16 f2bf(float f) {
    union { float f; unsigned u; } x; x.f = f;
    unsigned r = x.u + 0x7fffu + ((x.u >> 16) & 1u);
    return (u16)(r >> 16);
}

__device__ __forceinline__ bfx16 join8(bfx8 lo, bfx8 hi) {
    bfx16 r;
#pragma unroll
    for (int i = 0; i < 8; ++i) { r[i] = lo[i]; r[i + 8] = hi[i]; }
    return r;
}

// A-operand (16x32 bf16, interleaved halves: half0 K=0..7,16..23; half1 K=8..15,24..31)
__device__ __forceinline__ bfx16 load_a16(const u16* row, int kbase, int half) {
    bfx8 lo = *reinterpret_cast<const bfx8*>(row + kbase + half * 8);
    bfx8 hi = *reinterpret_cast<const bfx8*>(row + kbase + 16 + half * 8);
    return join8(lo, hi);
}
// B-operand (32x16 bf16, interleaved halves: half0 K=0..7,16..23; half1 K=8..15,24..31)
__device__ __forceinline__ bfx16 load_b16(const u16* row, int kbase, int half) {
    bfx8 lo = *reinterpret_cast<const bfx8*>(row + kbase + half * 8);
    bfx8 hi = *reinterpret_cast<const bfx8*>(row + kbase + 16 + half * 8);
    return join8(lo, hi);
}

__device__ __forceinline__ v8f wmma_bf16(bfx16 a, bfx16 b, v8f c) {
    return __builtin_amdgcn_wmma_f32_16x16x32_bf16(false, a, false, b, (short)0, c,
                                                   false, false);
}

// ---------------- 1) RMSNorm -> bf16 ----------------
__global__ void __launch_bounds__(256)
rmsnorm_kernel(const float* __restrict__ tokens, const float* __restrict__ w,
               u16* __restrict__ out) {
    const int row = blockIdx.x;            // 0..4095
    const int t   = threadIdx.x;           // 0..255
    __shared__ float red[256];
    const float* src = tokens + (size_t)row * DIMc;
    float xv[8];
    float s = 0.f;
#pragma unroll
    for (int i = 0; i < 8; ++i) { xv[i] = src[t * 8 + i]; s += xv[i] * xv[i]; }
    red[t] = s;
    __syncthreads();
    for (int o = 128; o > 0; o >>= 1) {
        if (t < o) red[t] += red[t + o];
        __syncthreads();
    }
    const float rs = rsqrtf(red[0] / (float)DIMc + 1.1920929e-07f);
    u16* dst = out + (size_t)row * DIMc;
#pragma unroll
    for (int i = 0; i < 8; ++i)
        dst[t * 8 + i] = f2bf(xv[i] * rs * w[t * 8 + i]);
}

// ---------------- 2) transpose + convert weights ----------------
__global__ void __launch_bounds__(256)
cvt_transpose(const float* __restrict__ W, u16* __restrict__ Wt, int K, int N) {
    int idx = blockIdx.x * 256 + threadIdx.x;
    if (idx < K * N) {
        int k = idx / N, n = idx % N;
        Wt[(size_t)n * K + k] = f2bf(W[idx]);
    }
}

// ---------------- 3) double-buffered bf16 WMMA GEMM ----------------
// C[MxN] = A[MxK] * Bt[NxK]^T ; block tile 128x64, wave tile 32x64
__global__ void __launch_bounds__(128)
gemm_bf16(const u16* __restrict__ A, const u16* __restrict__ Bt,
          float* __restrict__ C, int K, int ldc) {
    __shared__ __align__(16) u16 Atile[2][128 * 32];
    __shared__ __align__(16) u16 Btile[2][64 * 32];
    const int t    = threadIdx.x;
    const int wave = t >> 5;
    const int lane = t & 31;
    const int half = lane >> 4;
    const int l16  = lane & 15;
    const int m0 = blockIdx.y * 128, n0 = blockIdx.x * 64;

    // per-thread staging slots (16B chunks): 4 for A-tile (8 KiB), 2 for B-tile (4 KiB)
    const u16* ga[4]; int la[4];
#pragma unroll
    for (int i = 0; i < 4; ++i) {
        int c = t + i * 128, row = c >> 2, kc = (c & 3) * 8;
        ga[i] = A + (size_t)(m0 + row) * K + kc;
        la[i] = row * 32 + kc;
    }
    const u16* gb[2]; int lb[2];
#pragma unroll
    for (int i = 0; i < 2; ++i) {
        int c = t + i * 128, row = c >> 2, kc = (c & 3) * 8;
        gb[i] = Bt + (size_t)(n0 + row) * K + kc;
        lb[i] = row * 32 + kc;
    }

    auto stage = [&](int buf, int kb) {
#if HAVE_ASYNC
#pragma unroll
        for (int i = 0; i < 4; ++i) async_ld_b128(ga[i] + kb, &Atile[buf][la[i]]);
#pragma unroll
        for (int i = 0; i < 2; ++i) async_ld_b128(gb[i] + kb, &Btile[buf][lb[i]]);
#else
#pragma unroll
        for (int i = 0; i < 4; ++i)
            *reinterpret_cast<uint4*>(&Atile[buf][la[i]]) =
                *reinterpret_cast<const uint4*>(ga[i] + kb);
#pragma unroll
        for (int i = 0; i < 2; ++i)
            *reinterpret_cast<uint4*>(&Btile[buf][lb[i]]) =
                *reinterpret_cast<const uint4*>(gb[i] + kb);
#endif
    };

    v8f acc[2][4] = {};

    stage(0, 0);
    wait_async();
    __syncthreads();

    int buf = 0;
    for (int kb = 0; kb < K; kb += 32) {
        if (kb + 32 < K) stage(buf ^ 1, kb + 32);   // prefetch next k-block

        bfx16 a0 = load_a16(&Atile[buf][(wave * 32 + l16) * 32], 0, half);
        bfx16 a1 = load_a16(&Atile[buf][(wave * 32 + 16 + l16) * 32], 0, half);
        bfx16 bfj[4];
#pragma unroll
        for (int j = 0; j < 4; ++j)
            bfj[j] = load_b16(&Btile[buf][(j * 16 + l16) * 32], 0, half);
#pragma unroll
        for (int j = 0; j < 4; ++j) {
            acc[0][j] = wmma_bf16(a0, bfj[j], acc[0][j]);
            acc[1][j] = wmma_bf16(a1, bfj[j], acc[1][j]);
        }

        wait_async();        // this wave's prefetch landed in LDS
        __syncthreads();     // everyone done reading buf / writing buf^1
        buf ^= 1;
    }

#pragma unroll
    for (int mi = 0; mi < 2; ++mi)
#pragma unroll
        for (int j = 0; j < 4; ++j)
#pragma unroll
            for (int r = 0; r < 8; ++r) {
                int m = m0 + wave * 32 + mi * 16 + half * 8 + r;
                C[(size_t)m * ldc + n0 + j * 16 + l16] = acc[mi][j][r];
            }
}

// ---------------- 4) head L2-norm + rotary -> bf16 [B][H][N][64] ------------
__global__ void __launch_bounds__(64)
headnorm_rope(const float* __restrict__ qkv, int basecol,
              const float* __restrict__ gamma, const float* __restrict__ rotary,
              u16* __restrict__ out, int nheads) {
    const int n = blockIdx.x, h = blockIdx.y, b = blockIdx.z, d = threadIdx.x;
    __shared__ float red[64];
    __shared__ float sh[64];
    float val = qkv[((size_t)(b * Nc + n)) * NQKV + basecol + h * DH + d];
    red[d] = val * val;
    __syncthreads();
    for (int s = 32; s > 0; s >>= 1) {
        if (d < s) red[d] += red[d + s];
        __syncthreads();
    }
    float denom = fmaxf(sqrtf(red[0]), 1e-12f);
    float scale = (gamma[h * DH + d] + 1.0f) * 8.0f;   // sqrt(64) = 8
    float xn = val / denom * scale;
    sh[d] = xn;
    __syncthreads();
    float rothalf = (d < 32) ? -sh[d + 32] : sh[d - 32];
    float rot = rotary[(size_t)n * DH + d];
    float o = xn * __cosf(rot) + rothalf * __sinf(rot);
    out[(((size_t)b * nheads + h) * Nc + n) * DH + d] = f2bf(o);
}

// ---------------- 5) V slice -> bf16 [B][KVH][N][64] ----------------
__global__ void __launch_bounds__(256)
prep_v(const float* __restrict__ qkv, u16* __restrict__ vout) {
    int idx = blockIdx.x * 256 + threadIdx.x;        // < 2^21
    int d = idx & 63, n = (idx >> 6) & 2047, h = (idx >> 17) & 7, b = idx >> 20;
    float val = qkv[((size_t)(b * Nc + n)) * NQKV + 2560 + h * DH + d];
    vout[(((size_t)b * KVH + h) * Nc + n) * DH + d] = f2bf(val);
}

// ---------------- 6) flash attention (GQA, non-causal) ----------------
__global__ void __launch_bounds__(128)
attn_kernel(const u16* __restrict__ Q, const u16* __restrict__ Kk,
            const u16* __restrict__ V, u16* __restrict__ O) {
    const int qh = blockIdx.y, b = blockIdx.z, kvh = qh >> 2;
    const int qbase = blockIdx.x * 64;
    const int t = threadIdx.x, wave = t >> 5, lane = t & 31;
    const int half = lane >> 4, l16 = lane & 15;

    __shared__ __align__(16) u16 Ktile[32 * 64];   // [key][d]
    __shared__ __align__(16) u16 Vt[64 * 32];      // [d][key]
    __shared__ __align__(16) u16 Ptile[4][16 * 32];

    const u16* qptr = Q + ((size_t)b * QH + qh) * Nc * DH;
    const u16* kptr = Kk + ((size_t)b * KVH + kvh) * Nc * DH;
    const u16* vptr = V + ((size_t)b * KVH + kvh) * Nc * DH;

    const int qrow = qbase + wave * 16 + l16;
    bfx16 qa0 = load_a16(qptr + (size_t)qrow * DH, 0, half);
    bfx16 qa1 = load_a16(qptr + (size_t)qrow * DH, 32, half);

    // K-tile staging slots: 2 x 16B per thread
    const int kc0 = t,       krow0 = kc0 >> 3, kd0 = (kc0 & 7) * 8;
    const int kc1 = t + 128, krow1 = kc1 >> 3, kd1 = (kc1 & 7) * 8;

    float mrow[8], lrow[8];
#pragma unroll
    for (int r = 0; r < 8; ++r) { mrow[r] = -1e30f; lrow[r] = 0.f; }
    v8f Oacc[4] = {};

    for (int kb = 0; kb < Nc; kb += 32) {
#if HAVE_ASYNC
        async_ld_b128(&kptr[(size_t)(kb + krow0) * DH + kd0], &Ktile[krow0 * 64 + kd0]);
        async_ld_b128(&kptr[(size_t)(kb + krow1) * DH + kd1], &Ktile[krow1 * 64 + kd1]);
#else
        *reinterpret_cast<uint4*>(&Ktile[krow0 * 64 + kd0]) =
            *reinterpret_cast<const uint4*>(&kptr[(size_t)(kb + krow0) * DH + kd0]);
        *reinterpret_cast<uint4*>(&Ktile[krow1 * 64 + kd1]) =
            *reinterpret_cast<const uint4*>(&kptr[(size_t)(kb + krow1) * DH + kd1]);
#endif
        {   // load V rows, write transposed into Vt
            int key = t >> 2, ds0 = (t & 3) * 16;
            const u16* src = &vptr[(size_t)(kb + key) * DH + ds0];
            u16 tmp[16];
            *reinterpret_cast<uint4*>(&tmp[0]) = *reinterpret_cast<const uint4*>(&src[0]);
            *reinterpret_cast<uint4*>(&tmp[8]) = *reinterpret_cast<const uint4*>(&src[8]);
#pragma unroll
            for (int i = 0; i < 16; ++i) Vt[(ds0 + i) * 32 + key] = tmp[i];
        }
        wait_async();
        __syncthreads();

        // S = Q K^T for 16 q-rows x 32 keys (two 16-key halves)
        v8f s0 = {}, s1 = {};
        {
            bfx16 b00 = load_b16(&Ktile[(l16) * 64], 0, half);
            bfx16 b01 = load_b16(&Ktile[(l16) * 64], 32, half);
            bfx16 b10 = load_b16(&Ktile[(16 + l16) * 64], 0, half);
            bfx16 b11 = load_b16(&Ktile[(16 + l16) * 64], 32, half);
            s0 = wmma_bf16(qa0, b00, s0);
            s0 = wmma_bf16(qa1, b01, s0);
            s1 = wmma_bf16(qa0, b10, s1);
            s1 = wmma_bf16(qa1, b11, s1);
        }

        // online softmax over the 32-key block
#pragma unroll
        for (int r = 0; r < 8; ++r) {
            float sa = s0[r] * 0.125f;
            float sb = s1[r] * 0.125f;
            float mx = fmaxf(sa, sb);
#pragma unroll
            for (int off = 1; off <= 8; off <<= 1)
                mx = fmaxf(mx, __shfl_xor(mx, off, 32));
            float mnew = fmaxf(mrow[r], mx);
            float alpha = __expf(mrow[r] - mnew);
            mrow[r] = mnew;
            float p0 = __expf(sa - mnew);
            float p1 = __expf(sb - mnew);
            float ps = p0 + p1;
#pragma unroll
            for (int off = 1; off <= 8; off <<= 1)
                ps += __shfl_xor(ps, off, 32);
            lrow[r] = lrow[r] * alpha + ps;
#pragma unroll
            for (int c = 0; c < 4; ++c) Oacc[c][r] *= alpha;
            Ptile[wave][(half * 8 + r) * 32 + l16]      = f2bf(p0);
            Ptile[wave][(half * 8 + r) * 32 + 16 + l16] = f2bf(p1);
        }

        // O += P @ V  (per-wave LDS traffic stays in order within the wave)
        bfx16 pa = load_a16(&Ptile[wave][l16 * 32], 0, half);
#pragma unroll
        for (int c = 0; c < 4; ++c) {
            bfx16 vb = load_b16(&Vt[(c * 16 + l16) * 32], 0, half);
            Oacc[c] = wmma_bf16(pa, vb, Oacc[c]);
        }
        __syncthreads();
    }

    // epilogue: normalize and store bf16 into [B][N][QH*64]
#pragma unroll
    for (int r = 0; r < 8; ++r) {
        float inv = 1.0f / lrow[r];
        int m = qbase + wave * 16 + half * 8 + r;
        u16* dst = O + ((size_t)b * Nc + m) * (QH * DH) + qh * DH;
#pragma unroll
        for (int c = 0; c < 4; ++c)
            dst[c * 16 + l16] = f2bf(Oacc[c][r] * inv);
    }
}

// ---------------- launcher ----------------
extern "C" void kernel_launch(void* const* d_in, const int* in_sizes, int n_in,
                              void* d_out, int out_size, void* d_ws, size_t ws_size,
                              hipStream_t stream) {
    (void)in_sizes; (void)n_in; (void)out_size; (void)ws_size;
    const float* tokens  = (const float*)d_in[0];
    const float* rotary  = (const float*)d_in[1];
    const float* norm_w  = (const float*)d_in[2];
    const float* Wq      = (const float*)d_in[3];
    const float* Wkv     = (const float*)d_in[4];
    const float* Wo      = (const float*)d_in[5];
    const float* gamma_q = (const float*)d_in[6];
    const float* gamma_k = (const float*)d_in[7];

    char* ws = (char*)d_ws;
    u16*   x_bf   = (u16*)(ws);                         // 16 MiB
    u16*   WqkvT  = (u16*)(ws + (16ull  << 20));        // 12 MiB
    u16*   WoT    = (u16*)(ws + (28ull  << 20));        // 8 MiB
    float* qkv    = (float*)(ws + (36ull << 20));       // 48 MiB
    u16*   qb     = (u16*)(ws + (84ull  << 20));        // 16 MiB
    u16*   kbuf   = (u16*)(ws + (100ull << 20));        // 4 MiB
    u16*   vbuf   = (u16*)(ws + (104ull << 20));        // 4 MiB
    u16*   ab     = (u16*)(ws + (108ull << 20));        // 16 MiB

    rmsnorm_kernel<<<Mrows, 256, 0, stream>>>(tokens, norm_w, x_bf);
    cvt_transpose<<<(DIMc * 2048) / 256, 256, 0, stream>>>(Wq, WqkvT, DIMc, 2048);
    cvt_transpose<<<(DIMc * 1024) / 256, 256, 0, stream>>>(Wkv, WqkvT + (size_t)2048 * DIMc,
                                                           DIMc, 1024);
    cvt_transpose<<<(DIMc * DIMc) / 256, 256, 0, stream>>>(Wo, WoT, DIMc, DIMc);

    gemm_bf16<<<dim3(NQKV / 64, Mrows / 128), 128, 0, stream>>>(x_bf, WqkvT, qkv,
                                                                DIMc, NQKV);

    headnorm_rope<<<dim3(Nc, QH, Bc), 64, 0, stream>>>(qkv, 0, gamma_q, rotary, qb, QH);
    headnorm_rope<<<dim3(Nc, KVH, Bc), 64, 0, stream>>>(qkv, 2048, gamma_k, rotary, kbuf, KVH);
    prep_v<<<(Bc * KVH * Nc * DH) / 256, 256, 0, stream>>>(qkv, vbuf);

    attn_kernel<<<dim3(Nc / 64, QH, Bc), 128, 0, stream>>>(qb, kbuf, vbuf, ab);

    gemm_bf16<<<dim3(DIMc / 64, Mrows / 128), 128, 0, stream>>>(ab, WoT, (float*)d_out,
                                                                DIMc, DIMc);
}
